// GatedFreqCrossAttn3D_21861383536948
// MI455X (gfx1250) — matrix operand order repaired and checked by reading, not compiled, not run on hardware
//
#include <hip/hip_runtime.h>
#include <hip/hip_bf16.h>
#include <math.h>

typedef _Float16 f16;
typedef __attribute__((ext_vector_type(16))) _Float16 v16h_t;
typedef __attribute__((ext_vector_type(8)))  float    v8f_t;

#define VOX   27000
#define CCH   64
#define NHI   7
#define NTOK  216
#define NTOKP 224
#define NWIN  125
#define GEMM_BLOCKS 211   /* ceil(27000/128) */

__device__ __constant__ float CS6[6] = {1.f, 0.5f, -0.5f, -1.f, -0.5f, 0.5f};
__device__ __constant__ float SN6[6] = {0.f, 0.8660254037844386f, 0.8660254037844386f, 0.f,
                                        -0.8660254037844386f, -0.8660254037844386f};
__device__ __constant__ float FD6[6] = {0.f, 1.f/6.f, 1.f/3.f, 0.5f, 1.f/3.f, 1.f/6.f};
__device__ __constant__ float FW4[4] = {0.f, 1.f/6.f, 1.f/3.f, 0.5f};

__device__ __forceinline__ int apack_pos(int m_local, int k_local) {
  const int lane = m_local + 16 * ((k_local >> 3) & 1);
  const int half = ((k_local >> 4) << 3) | (k_local & 7);
  return lane * 16 + half;
}

__device__ __forceinline__ v8f_t wmma_f16(v16h_t a, v16h_t b, v8f_t c) {
  return __builtin_amdgcn_wmma_f32_16x16x32_f16(false, a, false, b, (short)0, c, false, false);
}

__device__ __forceinline__ long win_voxel(int wd, int wh, int ww, int t) {
  const int z = t / 36, y = (t / 6) % 6, x = t % 6;
  return ((long)((wd * 6 + z) * 30 + (wh * 6 + y))) * 30 + (ww * 6 + x);
}

__device__ void pack64g(f16* dst, const float* A, int rowstride, int coloff,
                        const float* scale, int tid) {
  for (int idx = tid; idx < 4096; idx += 256) {
    const int o = idx >> 6, c = idx & 63;
    float v = A[o * rowstride + coloff + c];
    if (scale) v *= scale[c];
    dst[(((c >> 5) * 4 + (o >> 4)) << 9) + apack_pos(o & 15, c & 31)] = (f16)v;
  }
}

// ---------------------------------------------------------------------------
// Prep: fold depthwise conv + embeds into GEMM weights; pack all A matrices
// into native wave32 WMMA A-fragment layout.
// ---------------------------------------------------------------------------
__global__ __launch_bounds__(256) void prep_kernel(
    const float* __restrict__ fuse_w, const float* __restrict__ hf_embed,
    const float* __restrict__ proj_dw_w, const float* __restrict__ proj_dw_b,
    const float* __restrict__ proj_pw_w, const float* __restrict__ proj_pw_b,
    const float* __restrict__ qh_dw_w, const float* __restrict__ qh_dw_b,
    const float* __restrict__ qh_pw_w, const float* __restrict__ qh_pw_b,
    const float* __restrict__ al_proj_w, const float* __restrict__ ah_proj_w,
    const float* __restrict__ gl_w, const float* __restrict__ gh_w, const float* __restrict__ gh_b,
    f16* fusepack, float* bias_fuse,
    f16* projpack, float* proj_bias,
    f16* qhpack, float* qh_bias,
    f16* alpack, f16* ahpack,
    f16* glp1, f16* glp2, f16* ghp1, f16* ghp2, float* ghbias)
{
  const int tid = threadIdx.x;
  // fuse: A'[o][k'=c*7+i] = fuse_w[o][i*64+c]  (matches highs' row order)
  for (int idx = tid; idx < 64 * 448; idx += 256) {
    const int o = idx / 448, kp = idx % 448;
    const int c = kp / 7, i = kp % 7;
    fusepack[(((kp >> 5) * 4 + (o >> 4)) << 9) + apack_pos(o & 15, kp & 31)] =
        (f16)fuse_w[o * 448 + i * 64 + c];
  }
  if (tid < 64) {
    float s = 0.f;
    for (int i = 0; i < 7; ++i)
      for (int c = 0; c < 64; ++c)
        s += fuse_w[tid * 448 + i * 64 + c] * hf_embed[i * 64 + c];
    bias_fuse[tid] = s;
  }
  for (int j = 0; j < 5; ++j)
    pack64g(projpack + j * 4096, proj_pw_w + j * 4096, 64, 0, proj_dw_w + j * 64, tid);
  for (int t = tid; t < 5 * 64; t += 256) {
    const int j = t / 64, o = t % 64;
    float s = proj_pw_b[j * 64 + o];
    for (int c = 0; c < 64; ++c)
      s += proj_pw_w[(j * 64 + o) * 64 + c] * proj_dw_b[j * 64 + c];
    proj_bias[t] = s;
  }
  for (int i = 0; i < 7; ++i)
    pack64g(qhpack + i * 4096, qh_pw_w + i * 4096, 64, 0, qh_dw_w + i * 64, tid);
  for (int t = tid; t < 7 * 64; t += 256) {
    const int i = t / 64, o = t % 64;
    float s = qh_pw_b[i * 64 + o];
    for (int c = 0; c < 64; ++c)
      s += qh_pw_w[(i * 64 + o) * 64 + c] *
           (qh_dw_w[i * 64 + c] * hf_embed[i * 64 + c] + qh_dw_b[i * 64 + c]);
    qh_bias[t] = s;
  }
  pack64g(alpack, al_proj_w, 64, 0, nullptr, tid);
  for (int i = 0; i < 7; ++i)
    pack64g(ahpack + i * 4096, ah_proj_w + i * 4096, 64, 0, nullptr, tid);
  pack64g(glp1, gl_w, 128, 0,  nullptr, tid);
  pack64g(glp2, gl_w, 128, 64, nullptr, tid);
  pack64g(ghp1, gh_w, 128, 0,  nullptr, tid);
  pack64g(ghp2, gh_w, 128, 64, nullptr, tid);
  for (int t = tid; t < 7 * 64; t += 256) {
    const int i = t / 64, o = t % 64;
    float s = gh_b[o];
    for (int c = 0; c < 64; ++c) s += gh_w[o * 128 + c] * hf_embed[i * 64 + c];
    ghbias[t] = s;
  }
}

// ---------------------------------------------------------------------------
// FFT enhancement: one block per 6^3 window.
// ---------------------------------------------------------------------------
__global__ __launch_bounds__(256) void fftenh_kernel(
    const float* __restrict__ low, const float* __restrict__ gate, const float* __restrict__ unc,
    const float* __restrict__ mu_raw, const float* __restrict__ sigma_raw,
    const float* __restrict__ gamma_p,
    const float* __restrict__ t_w1, const float* __restrict__ t_b1,
    const float* __restrict__ t_w2, const float* __restrict__ t_b2,
    const float* __restrict__ hb_w, const float* __restrict__ hb_b,
    const float* __restrict__ hs_w, const float* __restrict__ hs_b,
    float* __restrict__ low_e)
{
  __shared__ float cond[193];
  __shared__ float h1s[32], h2s[32], wsm[8];
  __shared__ float sstr;
  __shared__ float sM[144];
  __shared__ float xin[16 * 216];
  __shared__ float bufA[16 * 288];
  __shared__ float bufB[16 * 288];

  const int tid = threadIdx.x;
  const int wi = blockIdx.x;
  const int wd = wi / 25, wh = (wi / 5) % 5, ww = wi % 5;

  if (tid < 64) {
    float gs = 0.f, ls = 0.f, l2 = 0.f;
    for (int t = 0; t < 216; ++t) {
      const long v = win_voxel(wd, wh, ww, t);
      gs += gate[(long)tid * VOX + v];
      const float lv = low[(long)tid * VOX + v];
      ls += lv; l2 += lv * lv;
    }
    const float lm = ls * (1.f / 216.f);
    cond[tid] = gs * (1.f / 216.f);
    cond[64 + tid] = lm;
    cond[128 + tid] = sqrtf(fmaxf(l2 * (1.f / 216.f) - lm * lm, 1e-8f));
  } else if (tid == 64) {
    float us = 0.f;
    for (int t = 0; t < 216; ++t) {
      float u = unc[win_voxel(wd, wh, ww, t)];
      us += fminf(fmaxf(u, 0.f), 1.f);
    }
    cond[192] = us * (1.f / 216.f);
  }
  __syncthreads();
  if (tid < 32) {
    float a = t_b1[tid];
    for (int i = 0; i < 193; ++i) a += t_w1[tid * 193 + i] * cond[i];
    h1s[tid] = 0.5f * a * (1.f + erff(a * 0.70710678f));
  }
  __syncthreads();
  if (tid < 32) {
    float a = t_b2[tid];
    for (int i = 0; i < 32; ++i) a += t_w2[tid * 32 + i] * h1s[i];
    h2s[tid] = 0.5f * a * (1.f + erff(a * 0.70710678f));
  }
  __syncthreads();
  if (tid < 8) {
    float a = hb_b[tid];
    for (int i = 0; i < 32; ++i) a += hb_w[tid * 32 + i] * h2s[i];
    wsm[tid] = a;
  } else if (tid == 32) {
    float a = hs_b[0];
    for (int i = 0; i < 32; ++i) a += hs_w[i] * h2s[i];
    sstr = 1.f / (1.f + __expf(-a));
  }
  __syncthreads();
  if (tid == 0) {
    float mx = wsm[0];
    for (int i = 1; i < 8; ++i) mx = fmaxf(mx, wsm[i]);
    float s = 0.f;
    for (int i = 0; i < 8; ++i) { wsm[i] = __expf(wsm[i] - mx); s += wsm[i]; }
    const float inv = 1.f / s;
    for (int i = 0; i < 8; ++i) wsm[i] *= inv;
  }
  __syncthreads();
  if (tid < 144) {
    const int d = tid / 24, h = (tid / 4) % 6, w = tid % 4;
    const float rr = sqrtf(FD6[d]*FD6[d] + FD6[h]*FD6[h] + FW4[w]*FW4[w]) /
                     (0.8660254037844386f + 1e-8f);
    float b[8]; float bsum = 0.f;
    for (int k = 0; k < 8; ++k) {
      const float mu = 1.f / (1.f + __expf(-mu_raw[k]));
      const float sg = log1pf(__expf(sigma_raw[k])) + 0.02f;
      const float zf = (rr - mu) / (sg + 1e-8f);
      b[k] = __expf(-0.5f * zf * zf);
      bsum += b[k];
    }
    const float inv = 1.f / fmaxf(bsum, 1e-8f);
    float M = 0.f;
    for (int k = 0; k < 8; ++k) M += wsm[k] * b[k] * inv;
    sM[tid] = 1.f + tanhf(gamma_p[0]) * sstr * M;
  }
  __syncthreads();

  for (int ch0 = 0; ch0 < 64; ch0 += 16) {
    for (int idx = tid; idx < 16 * 216; idx += 256) {
      const int c = idx / 216, t = idx % 216;
      xin[idx] = low[(long)(ch0 + c) * VOX + win_voxel(wd, wh, ww, t)];
    }
    __syncthreads();
    // forward rfft along x
    for (int idx = tid; idx < 576; idx += 256) {
      const int c = idx / 36, d = (idx % 36) / 6, h = idx % 6;
      const float* src = &xin[c * 216 + (d * 6 + h) * 6];
      float* dst = &bufA[((c * 6 + d) * 6 + h) * 8];
      for (int w = 0; w < 4; ++w) {
        float re = 0.f, im = 0.f;
        for (int x = 0; x < 6; ++x) {
          const int k = (w * x) % 6;
          re += src[x] * CS6[k];
          im -= src[x] * SN6[k];
        }
        dst[w * 2] = re; dst[w * 2 + 1] = im;
      }
    }
    __syncthreads();
    // forward DFT along h
    for (int idx = tid; idx < 384; idx += 256) {
      const int c = idx / 24, d = (idx % 24) / 4, w = idx % 4;
      float ar[6], ai[6];
      for (int h = 0; h < 6; ++h) {
        ar[h] = bufA[((c * 6 + d) * 6 + h) * 8 + w * 2];
        ai[h] = bufA[((c * 6 + d) * 6 + h) * 8 + w * 2 + 1];
      }
      for (int ho = 0; ho < 6; ++ho) {
        float re = 0.f, im = 0.f;
        for (int h = 0; h < 6; ++h) {
          const int k = (ho * h) % 6;
          re += ar[h] * CS6[k] + ai[h] * SN6[k];
          im += ai[h] * CS6[k] - ar[h] * SN6[k];
        }
        bufB[((c * 6 + d) * 6 + ho) * 8 + w * 2] = re;
        bufB[((c * 6 + d) * 6 + ho) * 8 + w * 2 + 1] = im;
      }
    }
    __syncthreads();
    // forward DFT along d
    for (int idx = tid; idx < 384; idx += 256) {
      const int c = idx / 24, h = (idx % 24) / 4, w = idx % 4;
      float ar[6], ai[6];
      for (int d = 0; d < 6; ++d) {
        ar[d] = bufB[((c * 6 + d) * 6 + h) * 8 + w * 2];
        ai[d] = bufB[((c * 6 + d) * 6 + h) * 8 + w * 2 + 1];
      }
      for (int dd = 0; dd < 6; ++dd) {
        float re = 0.f, im = 0.f;
        for (int d = 0; d < 6; ++d) {
          const int k = (dd * d) % 6;
          re += ar[d] * CS6[k] + ai[d] * SN6[k];
          im += ai[d] * CS6[k] - ar[d] * SN6[k];
        }
        bufA[((c * 6 + dd) * 6 + h) * 8 + w * 2] = re;
        bufA[((c * 6 + dd) * 6 + h) * 8 + w * 2 + 1] = im;
      }
    }
    __syncthreads();
    for (int idx = tid; idx < 16 * 144; idx += 256) {
      const int c = idx / 144, r = idx % 144;
      bufA[c * 288 + r * 2] *= sM[r];
      bufA[c * 288 + r * 2 + 1] *= sM[r];
    }
    __syncthreads();
    // inverse DFT along d
    for (int idx = tid; idx < 384; idx += 256) {
      const int c = idx / 24, h = (idx % 24) / 4, w = idx % 4;
      float ar[6], ai[6];
      for (int d = 0; d < 6; ++d) {
        ar[d] = bufA[((c * 6 + d) * 6 + h) * 8 + w * 2];
        ai[d] = bufA[((c * 6 + d) * 6 + h) * 8 + w * 2 + 1];
      }
      for (int dd = 0; dd < 6; ++dd) {
        float re = 0.f, im = 0.f;
        for (int d = 0; d < 6; ++d) {
          const int k = (dd * d) % 6;
          re += ar[d] * CS6[k] - ai[d] * SN6[k];
          im += ai[d] * CS6[k] + ar[d] * SN6[k];
        }
        bufB[((c * 6 + dd) * 6 + h) * 8 + w * 2] = re;
        bufB[((c * 6 + dd) * 6 + h) * 8 + w * 2 + 1] = im;
      }
    }
    __syncthreads();
    // inverse DFT along h
    for (int idx = tid; idx < 384; idx += 256) {
      const int c = idx / 24, d = (idx % 24) / 4, w = idx % 4;
      float ar[6], ai[6];
      for (int h = 0; h < 6; ++h) {
        ar[h] = bufB[((c * 6 + d) * 6 + h) * 8 + w * 2];
        ai[h] = bufB[((c * 6 + d) * 6 + h) * 8 + w * 2 + 1];
      }
      for (int ho = 0; ho < 6; ++ho) {
        float re = 0.f, im = 0.f;
        for (int h = 0; h < 6; ++h) {
          const int k = (ho * h) % 6;
          re += ar[h] * CS6[k] - ai[h] * SN6[k];
          im += ai[h] * CS6[k] + ar[h] * SN6[k];
        }
        bufA[((c * 6 + d) * 6 + ho) * 8 + w * 2] = re;
        bufA[((c * 6 + d) * 6 + ho) * 8 + w * 2 + 1] = im;
      }
    }
    __syncthreads();
    // inverse rfft along x + store (overall 1/216 ortho-ortho norm)
    for (int idx = tid; idx < 576; idx += 256) {
      const int c = idx / 36, d = (idx % 36) / 6, h = idx % 6;
      const float* s = &bufA[((c * 6 + d) * 6 + h) * 8];
      const float a0 = s[0], a3 = s[6];
      for (int x = 0; x < 6; ++x) {
        float y = a0 + ((x & 1) ? -a3 : a3);
        for (int w = 1; w <= 2; ++w) {
          const int k = (w * x) % 6;
          y += 2.f * (s[w * 2] * CS6[k] - s[w * 2 + 1] * SN6[k]);
        }
        const int t = (d * 6 + h) * 6 + x;
        low_e[(long)(ch0 + c) * VOX + win_voxel(wd, wh, ww, t)] = y * (1.f / 216.f);
      }
    }
    __syncthreads();
  }
}

// ---------------------------------------------------------------------------
// Stage one 32 x 128 f32 chunk of X into LDS as packed f16 WMMA B fragments.
// Fast path: 4x float4 (global_load_b128) per thread, single wait.
// ---------------------------------------------------------------------------
__device__ __forceinline__ void stage_b_chunk(
    f16 (*bsh)[32][16], const float* __restrict__ X, long xs,
    int kk, int vbase, int tid)
{
  const int krow = tid >> 3;            // 0..31
  const int nb   = (tid & 7) * 16;
  const int lane_hi = ((krow >> 3) & 1) ? 16 : 0;
  const int khalf = ((krow >> 4) << 3) | (krow & 7);
  const int sub = tid & 7;
  const float* xr = X + (long)(kk + krow) * xs;
  if (vbase + 128 <= VOX) {
    const float4* xr4 = (const float4*)(xr + vbase + nb);
    const float4 r0 = xr4[0], r1 = xr4[1], r2 = xr4[2], r3 = xr4[3];
    f16* dst = &bsh[sub][lane_hi][khalf];
    dst[0 * 16]  = (f16)r0.x; dst[1 * 16]  = (f16)r0.y;
    dst[2 * 16]  = (f16)r0.z; dst[3 * 16]  = (f16)r0.w;
    dst[4 * 16]  = (f16)r1.x; dst[5 * 16]  = (f16)r1.y;
    dst[6 * 16]  = (f16)r1.z; dst[7 * 16]  = (f16)r1.w;
    dst[8 * 16]  = (f16)r2.x; dst[9 * 16]  = (f16)r2.y;
    dst[10 * 16] = (f16)r2.z; dst[11 * 16] = (f16)r2.w;
    dst[12 * 16] = (f16)r3.x; dst[13 * 16] = (f16)r3.y;
    dst[14 * 16] = (f16)r3.z; dst[15 * 16] = (f16)r3.w;
  } else {
#pragma unroll
    for (int j = 0; j < 16; ++j) {
      int v = vbase + nb + j;
      if (v >= VOX) v = VOX - 1;
      bsh[sub][j + lane_hi][khalf] = (f16)xr[v];
    }
  }
}

// ---------------------------------------------------------------------------
// WMMA GEMM: Y(64 x 27000) = A(64 x K, packed f16) * X(K x 27000) + bias
// ---------------------------------------------------------------------------
__global__ __launch_bounds__(256) void gemm_plain_kernel(
    const f16* __restrict__ Apack, const float* __restrict__ bias,
    const float* __restrict__ X, long xs,
    float* __restrict__ Y, long ys, int K)
{
  __shared__ __attribute__((aligned(32))) f16 bsh[8][32][16];
  const int tid = threadIdx.x;
  const int wv = tid >> 5, lane = tid & 31;
  const int mtile = wv & 3, nset = wv >> 2;
  const int vbase = blockIdx.x * 128;
  const v8f_t vz = {0.f,0.f,0.f,0.f,0.f,0.f,0.f,0.f};
  v8f_t acc[4] = {vz, vz, vz, vz};

  for (int kk = 0; kk < K; kk += 32) {
    stage_b_chunk(bsh, X, xs, kk, vbase, tid);
    if (kk + 32 < K)
      __builtin_prefetch(X + (long)(kk + 32 + (tid >> 3)) * xs + vbase + (tid & 7) * 16, 0, 1);
    __syncthreads();
    const v16h_t a = *((const v16h_t*)(Apack + (((kk >> 5) * 4 + mtile) << 9) + lane * 16));
#pragma unroll
    for (int t = 0; t < 4; ++t) {
      const v16h_t b = *((const v16h_t*)&bsh[nset + 2 * t][lane][0]);
      acc[t] = wmma_f16(a, b, acc[t]);
    }
    __syncthreads();
  }
  const int hl = lane >> 4, nlo = lane & 15;
#pragma unroll
  for (int t = 0; t < 4; ++t) {
    const int n = vbase + (nset + 2 * t) * 16 + nlo;
    if (n < VOX) {
#pragma unroll
      for (int r = 0; r < 8; ++r) {
        const int m = mtile * 16 + r + 8 * hl;
        Y[(long)m * ys + n] = acc[t][r] + bias[m];
      }
    }
  }
}

// ---------------------------------------------------------------------------
// K=128 gate GEMM + fused sigmoid gate / residual combine epilogue.
// ---------------------------------------------------------------------------
__global__ __launch_bounds__(256) void gemm_gate_combine_kernel(
    const f16* __restrict__ A1, const f16* __restrict__ A2, const float* __restrict__ gbias,
    const float* __restrict__ X1, long s1, const float* __restrict__ X2, long s2,
    const float* __restrict__ ebias, const float* __restrict__ base, long sb,
    const float* __restrict__ delta, float* __restrict__ out, long so,
    const float* __restrict__ beta)
{
  __shared__ __attribute__((aligned(32))) f16 bsh[8][32][16];
  const int tid = threadIdx.x;
  const int wv = tid >> 5, lane = tid & 31;
  const int mtile = wv & 3, nset = wv >> 2;
  const int vbase = blockIdx.x * 128;
  const v8f_t vz = {0.f,0.f,0.f,0.f,0.f,0.f,0.f,0.f};
  v8f_t acc[4] = {vz, vz, vz, vz};

  for (int p = 0; p < 2; ++p) {
    const float* X = p ? X2 : X1;
    const long xs = p ? s2 : s1;
    const f16* Ap = p ? A2 : A1;
    for (int kk = 0; kk < 64; kk += 32) {
      stage_b_chunk(bsh, X, xs, kk, vbase, tid);
      __syncthreads();
      const v16h_t a = *((const v16h_t*)(Ap + (((kk >> 5) * 4 + mtile) << 9) + lane * 16));
#pragma unroll
      for (int t = 0; t < 4; ++t) {
        const v16h_t b = *((const v16h_t*)&bsh[nset + 2 * t][lane][0]);
        acc[t] = wmma_f16(a, b, acc[t]);
      }
      __syncthreads();
    }
  }
  const float scale = beta ? tanhf(beta[0]) : 1.f;
  const int hl = lane >> 4, nlo = lane & 15;
#pragma unroll
  for (int t = 0; t < 4; ++t) {
    const int n = vbase + (nset + 2 * t) * 16 + nlo;
    if (n < VOX) {
#pragma unroll
      for (int r = 0; r < 8; ++r) {
        const int m = mtile * 16 + r + 8 * hl;
        const float g = 1.f / (1.f + __expf(-(acc[t][r] + gbias[m])));
        const float bse = base[(long)m * sb + n] + (ebias ? ebias[m] : 0.f);
        out[(long)m * so + n] = bse + scale * g * delta[(long)m * VOX + n];
      }
    }
  }
}

// ---------------------------------------------------------------------------
// Windowed cross-attention: one block per window, 8 heads x 14 row-strips
// as wave tasks; WMMA for QK^T and PV, lane-shuffle softmax.
// ---------------------------------------------------------------------------
__global__ __launch_bounds__(256) void attn_kernel(
    const float* __restrict__ Q, const float* __restrict__ Kx, const float* __restrict__ Vx,
    const float* __restrict__ qg, const float* __restrict__ qb,
    const float* __restrict__ kvg, const float* __restrict__ kvb,
    float* __restrict__ O)
{
  __shared__ __attribute__((aligned(32))) f16 qn[NTOKP][CCH];
  __shared__ __attribute__((aligned(32))) f16 kn[NTOKP][CCH];
  __shared__ __attribute__((aligned(32))) f16 vn[NTOKP][CCH];
  __shared__ __attribute__((aligned(32))) f16 pbuf[8][16][NTOKP];

  const int tid = threadIdx.x;
  const int wv = tid >> 5, lane = tid & 31;
  const int wi = blockIdx.x;
  const int wd = wi / 25, wh = (wi / 5) % 5, ww = wi % 5;

  for (int t = tid; t < NTOKP; t += 256) {
    if (t < NTOK) {
      const long v = win_voxel(wd, wh, ww, t);
      float buf[CCH];
      float mean = 0.f;
      for (int c = 0; c < CCH; ++c) { const float x = Q[(long)c * VOX + v]; buf[c] = x; mean += x; }
      mean *= (1.f / 64.f);
      float var = 0.f;
      for (int c = 0; c < CCH; ++c) { const float d = buf[c] - mean; var += d * d; }
      float inv = rsqrtf(var * (1.f / 64.f) + 1e-5f);
      for (int c = 0; c < CCH; ++c)
        qn[t][c] = (f16)(((buf[c] - mean) * inv * qg[c] + qb[c]) * 0.35355339059327373f);
      mean = 0.f;
      for (int c = 0; c < CCH; ++c) { const float x = Kx[(long)c * VOX + v]; buf[c] = x; mean += x; }
      mean *= (1.f / 64.f);
      var = 0.f;
      for (int c = 0; c < CCH; ++c) { const float d = buf[c] - mean; var += d * d; }
      inv = rsqrtf(var * (1.f / 64.f) + 1e-5f);
      for (int c = 0; c < CCH; ++c)
        kn[t][c] = (f16)((buf[c] - mean) * inv * kvg[c] + kvb[c]);
      mean = 0.f;
      for (int c = 0; c < CCH; ++c) { const float x = Vx[(long)c * VOX + v]; buf[c] = x; mean += x; }
      mean *= (1.f / 64.f);
      var = 0.f;
      for (int c = 0; c < CCH; ++c) { const float d = buf[c] - mean; var += d * d; }
      inv = rsqrtf(var * (1.f / 64.f) + 1e-5f);
      for (int c = 0; c < CCH; ++c)
        vn[t][c] = (f16)((buf[c] - mean) * inv * kvg[c] + kvb[c]);
    } else {
      for (int c = 0; c < CCH; ++c) {
        qn[t][c] = (f16)0.f; kn[t][c] = (f16)0.f; vn[t][c] = (f16)0.f;
      }
    }
  }
  __syncthreads();

  const v8f_t vz = {0.f,0.f,0.f,0.f,0.f,0.f,0.f,0.f};
  for (int task = wv; task < 112; task += 8) {
    const int head = task / 14, strip = task % 14;
    union { v16h_t v; f16 h[16]; } afrag;
    {
      const int row = strip * 16 + (lane & 15);
#pragma unroll
      for (int j = 0; j < 16; ++j) afrag.h[j] = (f16)0.f;
      if (lane < 16) {
#pragma unroll
        for (int j = 0; j < 8; ++j) afrag.h[j] = qn[row][head * 8 + j];
      }
    }
    v8f_t sc[14];
#pragma unroll
    for (int j = 0; j < 14; ++j) {
      union { v16h_t v; f16 h[16]; } bfrag;
#pragma unroll
      for (int q = 0; q < 16; ++q) bfrag.h[q] = (f16)0.f;
      if (lane < 16) {
        const int tk = j * 16 + lane;
#pragma unroll
        for (int q = 0; q < 8; ++q) bfrag.h[q] = kn[tk][head * 8 + q];
      }
      sc[j] = wmma_f16(afrag.v, bfrag.v, vz);
    }
    if ((lane & 15) >= 8) {           // key tokens 216..223 are padding
#pragma unroll
      for (int r = 0; r < 8; ++r) sc[13][r] = -1e30f;
    }
#pragma unroll
    for (int r = 0; r < 8; ++r) {
      float mx = -1e30f;
#pragma unroll
      for (int j = 0; j < 14; ++j) mx = fmaxf(mx, sc[j][r]);
      for (int off = 1; off < 16; off <<= 1) mx = fmaxf(mx, __shfl_xor(mx, off, 32));
      float ssum = 0.f;
#pragma unroll
      for (int j = 0; j < 14; ++j) { const float e = __expf(sc[j][r] - mx); sc[j][r] = e; ssum += e; }
      for (int off = 1; off < 16; off <<= 1) ssum += __shfl_xor(ssum, off, 32);
      const float inv = 1.f / ssum;
#pragma unroll
      for (int j = 0; j < 14; ++j) sc[j][r] *= inv;
    }
    // re-fragment P (C-layout -> A-layout) through wave-private LDS
    const int rbase = (lane >= 16) ? 8 : 0;
#pragma unroll
    for (int j = 0; j < 14; ++j) {
#pragma unroll
      for (int r = 0; r < 8; ++r)
        pbuf[wv][rbase + r][j * 16 + (lane & 15)] = (f16)sc[j][r];
    }
    v8f_t oacc = vz;
#pragma unroll
    for (int kc = 0; kc < 7; ++kc) {
      union { v16h_t v; f16 h[16]; } a2, b2;
      const int rowa = lane & 15;
      const int klo = (lane >= 16) ? 8 : 0;
#pragma unroll
      for (int hh = 0; hh < 16; ++hh) {
        const int k = kc * 32 + 2 * klo + hh;
        a2.h[hh] = pbuf[wv][rowa][k];
      }
      const int nn = lane & 15;
      const int kbase = kc * 32 + ((lane >= 16) ? 16 : 0);
#pragma unroll
      for (int hh = 0; hh < 16; ++hh)
        b2.h[hh] = (nn < 8) ? vn[kbase + hh][head * 8 + nn] : (f16)0.f;
      oacc = wmma_f16(a2.v, b2.v, oacc);
    }
#pragma unroll
    for (int r = 0; r < 8; ++r) {
      const int tok = strip * 16 + r + rbase;
      const int dch = lane & 15;
      if (tok < NTOK && dch < 8)
        O[(long)(head * 8 + dch) * VOX + win_voxel(wd, wh, ww, tok)] = oacc[r];
    }
  }
}

// ---------------------------------------------------------------------------
extern "C" void kernel_launch(void* const* d_in, const int* in_sizes, int n_in,
                              void* d_out, int out_size, void* d_ws, size_t ws_size,
                              hipStream_t stream)
{
  (void)in_sizes; (void)n_in; (void)out_size; (void)ws_size;
  const float* low      = (const float*)d_in[0];
  const float* highs    = (const float*)d_in[1];
  const float* gate_dec = (const float*)d_in[2];
  const float* unc      = (const float*)d_in[3];
  const float* hf_embed = (const float*)d_in[4];
  const float* fuse_w   = (const float*)d_in[5];
  const float* proj_dw_w= (const float*)d_in[6];
  const float* proj_dw_b= (const float*)d_in[7];
  const float* proj_pw_w= (const float*)d_in[8];
  const float* proj_pw_b= (const float*)d_in[9];
  const float* qh_dw_w  = (const float*)d_in[10];
  const float* qh_dw_b  = (const float*)d_in[11];
  const float* qh_pw_w  = (const float*)d_in[12];
  const float* qh_pw_b  = (const float*)d_in[13];
  const float* al_nq_g  = (const float*)d_in[14];
  const float* al_nq_b  = (const float*)d_in[15];
  const float* al_nkv_g = (const float*)d_in[16];
  const float* al_nkv_b = (const float*)d_in[17];
  const float* al_proj_w= (const float*)d_in[18];
  const float* al_proj_b= (const float*)d_in[19];
  const float* ah_nq_g  = (const float*)d_in[20];
  const float* ah_nq_b  = (const float*)d_in[21];
  const float* ah_nkv_g = (const float*)d_in[22];
  const float* ah_nkv_b = (const float*)d_in[23];
  const float* ah_proj_w= (const float*)d_in[24];
  const float* ah_proj_b= (const float*)d_in[25];
  const float* gl_w     = (const float*)d_in[26];
  const float* gl_b     = (const float*)d_in[27];
  const float* gh_w     = (const float*)d_in[28];
  const float* gh_b     = (const float*)d_in[29];
  const float* beta_hf  = (const float*)d_in[30];
  const float* mu_raw   = (const float*)d_in[31];
  const float* sigma_raw= (const float*)d_in[32];
  const float* gamma    = (const float*)d_in[33];
  const float* t_w1 = (const float*)d_in[34];
  const float* t_b1 = (const float*)d_in[35];
  const float* t_w2 = (const float*)d_in[36];
  const float* t_b2 = (const float*)d_in[37];
  const float* hb_w = (const float*)d_in[38];
  const float* hb_b = (const float*)d_in[39];
  const float* hs_w = (const float*)d_in[40];
  const float* hs_b = (const float*)d_in[41];

  char* wsp = (char*)d_ws;
  size_t off = 0;
  auto alloc = [&](size_t bytes) -> void* {
    void* p = wsp + off;
    off = (off + bytes + 255) & ~(size_t)255;
    return p;
  };
  const size_t FBUF = (size_t)CCH * VOX * sizeof(float);
  float* low_e = (float*)alloc(FBUF);
  float* fbuf  = (float*)alloc(FBUF);
  float* qlow  = (float*)alloc(FBUF);
  float* klow  = (float*)alloc(FBUF);
  float* vlow  = (float*)alloc(FBUF);
  float* kf    = (float*)alloc(FBUF);
  float* vf    = (float*)alloc(FBUF);
  float* qhi   = (float*)alloc(FBUF);
  float* obuf  = (float*)alloc(FBUF);
  float* delta = (float*)alloc(FBUF);
  f16*   fusepack  = (f16*)alloc(14 * 4 * 512 * sizeof(f16));
  float* bias_fuse = (float*)alloc(64 * sizeof(float));
  f16*   projpack  = (f16*)alloc(5 * 4096 * sizeof(f16));
  float* proj_bias = (float*)alloc(5 * 64 * sizeof(float));
  f16*   qhpack    = (f16*)alloc(7 * 4096 * sizeof(f16));
  float* qh_bias   = (float*)alloc(7 * 64 * sizeof(float));
  f16*   alpack    = (f16*)alloc(4096 * sizeof(f16));
  f16*   ahpack    = (f16*)alloc(7 * 4096 * sizeof(f16));
  f16*   glp1 = (f16*)alloc(4096 * sizeof(f16));
  f16*   glp2 = (f16*)alloc(4096 * sizeof(f16));
  f16*   ghp1 = (f16*)alloc(4096 * sizeof(f16));
  f16*   ghp2 = (f16*)alloc(4096 * sizeof(f16));
  float* ghbias = (float*)alloc(7 * 64 * sizeof(float));

  prep_kernel<<<1, 256, 0, stream>>>(
      fuse_w, hf_embed, proj_dw_w, proj_dw_b, proj_pw_w, proj_pw_b,
      qh_dw_w, qh_dw_b, qh_pw_w, qh_pw_b, al_proj_w, ah_proj_w, gl_w, gh_w, gh_b,
      fusepack, bias_fuse, projpack, proj_bias, qhpack, qh_bias, alpack, ahpack,
      glp1, glp2, ghp1, ghp2, ghbias);

  fftenh_kernel<<<NWIN, 256, 0, stream>>>(
      low, gate_dec, unc, mu_raw, sigma_raw, gamma,
      t_w1, t_b1, t_w2, t_b2, hb_w, hb_b, hs_w, hs_b, low_e);

  gemm_plain_kernel<<<GEMM_BLOCKS, 256, 0, stream>>>(
      fusepack, bias_fuse, highs, (long)VOX, fbuf, (long)VOX, 448);
  gemm_plain_kernel<<<GEMM_BLOCKS, 256, 0, stream>>>(
      projpack + 0 * 4096, proj_bias + 0 * 64, low_e, (long)VOX, qlow, (long)VOX, 64);
  gemm_plain_kernel<<<GEMM_BLOCKS, 256, 0, stream>>>(
      projpack + 1 * 4096, proj_bias + 1 * 64, low_e, (long)VOX, klow, (long)VOX, 64);
  gemm_plain_kernel<<<GEMM_BLOCKS, 256, 0, stream>>>(
      projpack + 2 * 4096, proj_bias + 2 * 64, low_e, (long)VOX, vlow, (long)VOX, 64);
  gemm_plain_kernel<<<GEMM_BLOCKS, 256, 0, stream>>>(
      projpack + 3 * 4096, proj_bias + 3 * 64, fbuf, (long)VOX, kf, (long)VOX, 64);
  gemm_plain_kernel<<<GEMM_BLOCKS, 256, 0, stream>>>(
      projpack + 4 * 4096, proj_bias + 4 * 64, fbuf, (long)VOX, vf, (long)VOX, 64);

  attn_kernel<<<NWIN, 256, 0, stream>>>(
      qlow, kf, vf, al_nq_g, al_nq_b, al_nkv_g, al_nkv_b, obuf);
  gemm_plain_kernel<<<GEMM_BLOCKS, 256, 0, stream>>>(
      alpack, al_proj_b, obuf, (long)VOX, delta, (long)VOX, 64);
  gemm_gate_combine_kernel<<<GEMM_BLOCKS, 256, 0, stream>>>(
      glp1, glp2, gl_b, low_e, (long)VOX, gate_dec, (long)VOX,
      nullptr, low_e, (long)VOX, delta, (float*)d_out, (long)VOX, nullptr);

  for (int i = 0; i < NHI; ++i) {
    gemm_plain_kernel<<<GEMM_BLOCKS, 256, 0, stream>>>(
        qhpack + i * 4096, qh_bias + i * 64,
        highs + (size_t)i * VOX, (long)NHI * VOX, qhi, (long)VOX, 64);
    attn_kernel<<<NWIN, 256, 0, stream>>>(
        qhi, klow, vlow, ah_nq_g + i * 64, ah_nq_b + i * 64,
        ah_nkv_g + i * 64, ah_nkv_b + i * 64, obuf);
    gemm_plain_kernel<<<GEMM_BLOCKS, 256, 0, stream>>>(
        ahpack + i * 4096, ah_proj_b + i * 64, obuf, (long)VOX, delta, (long)VOX, 64);
    gemm_gate_combine_kernel<<<GEMM_BLOCKS, 256, 0, stream>>>(
        ghp1, ghp2, ghbias + i * 64,
        highs + (size_t)i * VOX, (long)NHI * VOX, gate_dec, (long)VOX,
        hf_embed + i * 64, highs + (size_t)i * VOX, (long)NHI * VOX,
        delta, (float*)d_out + (size_t)CCH * VOX + (size_t)i * VOX, (long)NHI * VOX, beta_hf);
  }
}
